// ConvPoolBlock_72146860638652
// MI455X (gfx1250) — matrix operand order repaired and checked by reading, not compiled, not run on hardware
//
#include <hip/hip_runtime.h>
#include <hip/hip_bf16.h>

// ---------------- problem constants (match reference) ----------------
#define BGR    64
#define NPG    512
#define NODES  (BGR * NPG)        // 32768
#define DEG_E  16
#define EDGES  (NODES * DEG_E)    // 524288
#define DF     512                // D_IN == D_OUT
#define KKEEP  256                // ceil(0.5 * N)

typedef __bf16 bf16_t;
typedef bf16_t v8bf  __attribute__((ext_vector_type(8)));
typedef bf16_t v16bf __attribute__((ext_vector_type(16)));
typedef float  v8f   __attribute__((ext_vector_type(8)));

__device__ __forceinline__ unsigned short f2bf(float f) {
    unsigned int u = __float_as_uint(f);
    u += 0x7FFFu + ((u >> 16) & 1u);          // round-to-nearest-even
    return (unsigned short)(u >> 16);
}

// ---------------- degree count ----------------
__global__ void k_deg(const int* __restrict__ src, const int* __restrict__ dst,
                      unsigned* __restrict__ degOut, unsigned* __restrict__ degIn) {
    int e = blockIdx.x * blockDim.x + threadIdx.x;
    if (e < EDGES) {
        atomicAdd(&degOut[src[e]], 1u);
        atomicAdd(&degIn[dst[e]], 1u);
    }
}

// ---------------- degree -> norm ----------------
__global__ void k_norm(const unsigned* __restrict__ degOut, const unsigned* __restrict__ degIn,
                       float* __restrict__ normOut, float* __restrict__ normIn) {
    int i = blockIdx.x * blockDim.x + threadIdx.x;
    if (i < NODES) {
        normOut[i] = rsqrtf(fmaxf((float)degOut[i], 1.0f));
        normIn[i]  = rsqrtf(fmaxf((float)degIn[i],  1.0f));
    }
}

// ---------------- exclusive scan of 32768 in-degrees (1 block, 1024 thr) ----------------
__launch_bounds__(1024)
__global__ void k_scan(const unsigned* __restrict__ degIn, unsigned* __restrict__ rowStart) {
    __shared__ unsigned part[1024];
    int t = threadIdx.x;
    int base = t * 32;
    unsigned loc[32];
    unsigned s = 0;
    #pragma unroll
    for (int i = 0; i < 32; i++) { loc[i] = degIn[base + i]; s += loc[i]; }
    part[t] = s;
    __syncthreads();
    for (int off = 1; off < 1024; off <<= 1) {
        unsigned mine = part[t];
        unsigned add  = (t >= off) ? part[t - off] : 0u;
        __syncthreads();
        part[t] = mine + add;
        __syncthreads();
    }
    unsigned run = part[t] - s;               // exclusive offset of this chunk
    #pragma unroll
    for (int i = 0; i < 32; i++) { rowStart[base + i] = run; run += loc[i]; }
}

// ---------------- CSR scatter (edges sorted by dst) ----------------
__global__ void k_scatter(const int* __restrict__ src, const int* __restrict__ dst,
                          const unsigned* __restrict__ rowStart, unsigned* __restrict__ cursor,
                          unsigned* __restrict__ csrSrc) {
    int e = blockIdx.x * blockDim.x + threadIdx.x;
    if (e < EDGES) {
        int d = dst[e];
        unsigned p = atomicAdd(&cursor[d], 1u);
        csrSrc[rowStart[d] + p] = (unsigned)src[e];
    }
}

// ---------------- f32 -> bf16 conversion (4 elems/thread) ----------------
__global__ void k_cvtx(const float4* __restrict__ x, ushort4* __restrict__ out, int n4) {
    int i = blockIdx.x * blockDim.x + threadIdx.x;
    if (i < n4) {
        float4 v = x[i];
        ushort4 o;
        o.x = f2bf(v.x); o.y = f2bf(v.y); o.z = f2bf(v.z); o.w = f2bf(v.w);
        out[i] = o;
    }
}

// ---------------- transpose + convert weight: Wt[n,k] = bf16(W[k,n]) ----------------
__global__ void k_cvtwT(const float* __restrict__ W, unsigned short* __restrict__ Wt) {
    int i = blockIdx.x * blockDim.x + threadIdx.x;   // i < DF*DF
    int n = i >> 9, k = i & 511;
    Wt[n * DF + k] = f2bf(W[k * DF + n]);
}

__global__ void k_swsum(const float* __restrict__ a, const float* __restrict__ b,
                        float* __restrict__ o) {
    int i = blockIdx.x * blockDim.x + threadIdx.x;
    if (i < DF) o[i] = a[i] + b[i];
}

// ---------------- WMMA GEMM: P[m,n] = (A @ W) * normOut[m] ----------------
// A: [NODES, DF] bf16 row-major.  Wt: [DF(out n), DF(k)] bf16 (pre-transposed W).
// Block = 256 thr (8 waves); wave -> 16(M) x 64(N) strip; grid = (DF/64, NODES/128).
__launch_bounds__(256)
__global__ void k_gemm(const unsigned short* __restrict__ Ab,
                       const unsigned short* __restrict__ Wtb,
                       const float* __restrict__ normOut,
                       float* __restrict__ P) {
    const int tid   = threadIdx.x;
    const int w     = tid >> 5;
    const int l     = tid & 31;
    const int lMod  = l & 15;
    const int lHalf = l >> 4;
    const int tileN   = blockIdx.x * 64;
    const int rowBase = blockIdx.y * 128 + w * 16;

    const bf16_t* A  = (const bf16_t*)Ab;
    const bf16_t* Bt = (const bf16_t*)Wtb;

    // A layout (16-bit 16x32): lanes 0-15 hold K[0..15], lanes 16-31 K[16..31], contiguous
    const bf16_t* aPtr = A + (size_t)(rowBase + lMod) * DF + (lHalf << 4);
    // B layout (16-bit 32x16): lanes 0-15 hold K[0..15], lanes 16-31 K[16..31], contiguous
    const bf16_t* bp0 = Bt + (size_t)(tileN +  0 + lMod) * DF + (lHalf << 4);
    const bf16_t* bp1 = Bt + (size_t)(tileN + 16 + lMod) * DF + (lHalf << 4);
    const bf16_t* bp2 = Bt + (size_t)(tileN + 32 + lMod) * DF + (lHalf << 4);
    const bf16_t* bp3 = Bt + (size_t)(tileN + 48 + lMod) * DF + (lHalf << 4);

    v8f acc0 = {}, acc1 = {}, acc2 = {}, acc3 = {};

    for (int kk = 0; kk < DF; kk += 32) {
        union { v16bf v; v8bf h[2]; } a, b;
        a.h[0] = *(const v8bf*)(aPtr + kk);
        a.h[1] = *(const v8bf*)(aPtr + kk + 8);
        __builtin_prefetch(aPtr + kk + 32, 0, 0);   // global_prefetch_b8

        b.h[0] = *(const v8bf*)(bp0 + kk);
        b.h[1] = *(const v8bf*)(bp0 + kk + 8);
        acc0 = __builtin_amdgcn_wmma_f32_16x16x32_bf16(false, a.v, false, b.v, (short)0, acc0, false, false);
        b.h[0] = *(const v8bf*)(bp1 + kk);
        b.h[1] = *(const v8bf*)(bp1 + kk + 8);
        acc1 = __builtin_amdgcn_wmma_f32_16x16x32_bf16(false, a.v, false, b.v, (short)0, acc1, false, false);
        b.h[0] = *(const v8bf*)(bp2 + kk);
        b.h[1] = *(const v8bf*)(bp2 + kk + 8);
        acc2 = __builtin_amdgcn_wmma_f32_16x16x32_bf16(false, a.v, false, b.v, (short)0, acc2, false, false);
        b.h[0] = *(const v8bf*)(bp3 + kk);
        b.h[1] = *(const v8bf*)(bp3 + kk + 8);
        acc3 = __builtin_amdgcn_wmma_f32_16x16x32_bf16(false, a.v, false, b.v, (short)0, acc3, false, false);
    }

    // C/D layout: lane l, elem r -> m = r + 8*(l>=16), n = l&15
    const int mAdd = lHalf * 8;
    float nrm[8];
    #pragma unroll
    for (int r = 0; r < 8; r++) nrm[r] = normOut[rowBase + mAdd + r];
    #pragma unroll
    for (int r = 0; r < 8; r++) {
        float* o = P + (size_t)(rowBase + mAdd + r) * DF + tileN + lMod;
        o[0]  = acc0[r] * nrm[r];
        o[16] = acc1[r] * nrm[r];
        o[32] = acc2[r] * nrm[r];
        o[48] = acc3[r] * nrm[r];
    }
}

// ---------------- CSR row aggregation + bias + relu (one wave per node) ----------------
template<bool BF16OUT>
__launch_bounds__(256)
__global__ void k_agg(const float* __restrict__ P, const unsigned* __restrict__ rowStart,
                      const unsigned* __restrict__ degIn, const unsigned* __restrict__ csrSrc,
                      const float* __restrict__ normIn, const float* __restrict__ bias,
                      void* __restrict__ outp) {
    int wave = (int)((blockIdx.x * blockDim.x + threadIdx.x) >> 5);
    int l = threadIdx.x & 31;
    if (wave >= NODES) return;
    float4 acc[4];
    #pragma unroll
    for (int it = 0; it < 4; it++) acc[it] = make_float4(0.f, 0.f, 0.f, 0.f);

    unsigned start = rowStart[wave], cnt = degIn[wave];
    for (unsigned e = 0; e < cnt; e++) {
        const float4* row = (const float4*)(P + (size_t)csrSrc[start + e] * DF);
        #pragma unroll
        for (int it = 0; it < 4; it++) {
            float4 v = row[l + 32 * it];
            acc[it].x += v.x; acc[it].y += v.y; acc[it].z += v.z; acc[it].w += v.w;
        }
    }
    float ni = normIn[wave];
    #pragma unroll
    for (int it = 0; it < 4; it++) {
        int col = 4 * (l + 32 * it);
        float4 bb = *(const float4*)(bias + col);
        float4 h;
        h.x = fmaxf(acc[it].x * ni + bb.x, 0.f);
        h.y = fmaxf(acc[it].y * ni + bb.y, 0.f);
        h.z = fmaxf(acc[it].z * ni + bb.z, 0.f);
        h.w = fmaxf(acc[it].w * ni + bb.w, 0.f);
        if (BF16OUT) {
            ushort4 o;
            o.x = f2bf(h.x); o.y = f2bf(h.y); o.z = f2bf(h.z); o.w = f2bf(h.w);
            *(ushort4*)((unsigned short*)outp + (size_t)wave * DF + col) = o;
        } else {
            *(float4*)((float*)outp + (size_t)wave * DF + col) = h;
        }
    }
}

// ---------------- fused score GEMV: u[v] = (h2[v] . (sW1+sW2)) * normOut[v] ----------------
__launch_bounds__(256)
__global__ void k_gemv(const float* __restrict__ H2, const float* __restrict__ sw,
                       const float* __restrict__ normOut, float* __restrict__ u) {
    int wave = (int)((blockIdx.x * blockDim.x + threadIdx.x) >> 5);
    int l = threadIdx.x & 31;
    if (wave >= NODES) return;
    const float4* row = (const float4*)(H2 + (size_t)wave * DF);
    const float4* wv  = (const float4*)sw;
    float s = 0.f;
    #pragma unroll
    for (int it = 0; it < 4; it++) {
        float4 a = row[l + 32 * it], b = wv[l + 32 * it];
        s += a.x * b.x + a.y * b.y + a.z * b.z + a.w * b.w;
    }
    #pragma unroll
    for (int m = 16; m >= 1; m >>= 1) s += __shfl_xor(s, m, 32);
    if (l == 0) u[wave] = s * normOut[wave];
}

// ---------------- scalar edge aggregation for the score ----------------
__global__ void k_score(const float* __restrict__ u, const unsigned* __restrict__ rowStart,
                        const unsigned* __restrict__ degIn, const unsigned* __restrict__ csrSrc,
                        const float* __restrict__ normIn,
                        const float* __restrict__ sb1, const float* __restrict__ sb2,
                        float* __restrict__ score) {
    int v = blockIdx.x * blockDim.x + threadIdx.x;
    if (v < NODES) {
        unsigned st = rowStart[v], c = degIn[v];
        float s = 0.f;
        for (unsigned e = 0; e < c; e++) s += u[csrSrc[st + e]];
        score[v] = 0.5f * (s * normIn[v] + sb1[0] + sb2[0]);
    }
}

// ---------------- per-graph top-K via bitonic sort in LDS (desc, index tiebreak) ----------------
__launch_bounds__(512)
__global__ void k_topk(const float* __restrict__ score, int* __restrict__ topIdx) {
    __shared__ float s[NPG];
    __shared__ int   id[NPG];
    int b = blockIdx.x, t = threadIdx.x;
    s[t] = score[b * NPG + t];
    id[t] = t;
    __syncthreads();
    for (int k = 2; k <= NPG; k <<= 1) {
        for (int j = k >> 1; j > 0; j >>= 1) {
            int p = t ^ j;
            if (p > t) {
                float sa = s[t], sb_ = s[p];
                int   ia = id[t], ib = id[p];
                bool aFirst  = (sa > sb_) || (sa == sb_ && ia < ib);
                bool keepDesc = ((t & k) == 0);
                if (keepDesc ? !aFirst : aFirst) {
                    s[t] = sb_; s[p] = sa; id[t] = ib; id[p] = ia;
                }
            }
            __syncthreads();
        }
    }
    if (t < KKEEP) topIdx[b * KKEEP + t] = id[t];
}

// ---------------- gated gather: pooled = h2[perm] * tanh(score[perm]) ----------------
__launch_bounds__(256)
__global__ void k_pool(const float* __restrict__ H2, const float* __restrict__ score,
                       const int* __restrict__ topIdx, float* __restrict__ pooled) {
    int wave = (int)((blockIdx.x * blockDim.x + threadIdx.x) >> 5);
    int l = threadIdx.x & 31;
    if (wave >= BGR * KKEEP) return;
    int b = wave / KKEEP;
    int node = b * NPG + topIdx[wave];
    float g = tanhf(score[node]);
    const float4* rin  = (const float4*)(H2 + (size_t)node * DF);
    float4*       rout = (float4*)(pooled + (size_t)wave * DF);
    #pragma unroll
    for (int it = 0; it < 4; it++) {
        float4 v = rin[l + 32 * it];
        v.x *= g; v.y *= g; v.z *= g; v.w *= g;
        rout[l + 32 * it] = v;
    }
}

// ---------------- readout: per-graph max || sum over K rows ----------------
__launch_bounds__(512)
__global__ void k_readout(const float* __restrict__ pooled, float* __restrict__ gout) {
    int b = blockIdx.x, t = threadIdx.x;
    float mx = -INFINITY, sm = 0.f;
    const float* base = pooled + (size_t)b * KKEEP * DF + t;
    for (int r = 0; r < KKEEP; r++) {
        float v = base[(size_t)r * DF];
        mx = fmaxf(mx, v);
        sm += v;
    }
    gout[b * 2 * DF + t]      = mx;
    gout[b * 2 * DF + DF + t] = sm;
}

// =====================================================================
extern "C" void kernel_launch(void* const* d_in, const int* in_sizes, int n_in,
                              void* d_out, int out_size, void* d_ws, size_t ws_size,
                              hipStream_t stream) {
    const float* x   = (const float*)d_in[0];
    const int*   src = (const int*)d_in[1];
    const int*   dst = (const int*)d_in[2];
    const float* W1  = (const float*)d_in[3];
    const float* b1  = (const float*)d_in[4];
    const float* W2  = (const float*)d_in[5];
    const float* b2  = (const float*)d_in[6];
    const float* sW1 = (const float*)d_in[7];
    const float* sb1 = (const float*)d_in[8];
    const float* sW2 = (const float*)d_in[9];
    const float* sb2 = (const float*)d_in[10];

    float* pooled = (float*)d_out;                              // [B*K, DF]
    float* gout   = (float*)d_out + (size_t)BGR * KKEEP * DF;   // [B, 2*DF]

    // ---- workspace carve-up ----
    char* w = (char*)d_ws;
    auto alloc = [&](size_t bytes) -> void* {
        void* p = (void*)w;
        w += (bytes + 255) & ~(size_t)255;
        return p;
    };
    unsigned* degOut   = (unsigned*)alloc(NODES * 4);
    unsigned* degIn    = (unsigned*)alloc(NODES * 4);
    unsigned* rowStart = (unsigned*)alloc(NODES * 4);
    unsigned* cursor   = (unsigned*)alloc(NODES * 4);
    float*    normOut  = (float*)alloc(NODES * 4);
    float*    normIn   = (float*)alloc(NODES * 4);
    unsigned* csrSrc   = (unsigned*)alloc(EDGES * 4);
    float*    sWsum    = (float*)alloc(DF * 4);
    float*    score    = (float*)alloc(NODES * 4);
    float*    uvec     = (float*)alloc(NODES * 4);
    int*      topIdx   = (int*)alloc(BGR * KKEEP * 4);
    unsigned short* Wt1b = (unsigned short*)alloc((size_t)DF * DF * 2);
    unsigned short* Wt2b = (unsigned short*)alloc((size_t)DF * DF * 2);
    unsigned short* Xb   = (unsigned short*)alloc((size_t)NODES * DF * 2); // reused as H1b
    float*    P    = (float*)alloc((size_t)NODES * DF * 4);
    float*    H2   = (float*)alloc((size_t)NODES * DF * 4);

    // ---- zero counters ----
    hipMemsetAsync(degOut, 0, NODES * 4, stream);
    hipMemsetAsync(degIn,  0, NODES * 4, stream);
    hipMemsetAsync(cursor, 0, NODES * 4, stream);

    // ---- graph structure ----
    k_deg<<<EDGES / 256, 256, 0, stream>>>(src, dst, degOut, degIn);
    k_norm<<<NODES / 256, 256, 0, stream>>>(degOut, degIn, normOut, normIn);
    k_scan<<<1, 1024, 0, stream>>>(degIn, rowStart);
    k_scatter<<<EDGES / 256, 256, 0, stream>>>(src, dst, rowStart, cursor, csrSrc);

    // ---- precision conversion ----
    k_cvtx<<<(NODES * DF / 4) / 256, 256, 0, stream>>>((const float4*)x, (ushort4*)Xb, NODES * DF / 4);
    k_cvtwT<<<(DF * DF) / 256, 256, 0, stream>>>(W1, Wt1b);
    k_cvtwT<<<(DF * DF) / 256, 256, 0, stream>>>(W2, Wt2b);
    k_swsum<<<2, 256, 0, stream>>>(sW1, sW2, sWsum);

    dim3 gemmGrid(DF / 64, NODES / 128);

    // ---- conv1: P = (x@W1)*normOut ; H1 = relu(agg*normIn + b1) -> bf16 (reuse Xb) ----
    k_gemm<<<gemmGrid, 256, 0, stream>>>(Xb, Wt1b, normOut, P);
    k_agg<true><<<NODES / 8, 256, 0, stream>>>(P, rowStart, degIn, csrSrc, normIn, b1, (void*)Xb);

    // ---- conv2: P = (h1@W2)*normOut ; H2 = relu(agg*normIn + b2) -> f32 ----
    k_gemm<<<gemmGrid, 256, 0, stream>>>(Xb, Wt2b, normOut, P);
    k_agg<false><<<NODES / 8, 256, 0, stream>>>(P, rowStart, degIn, csrSrc, normIn, b2, (void*)H2);

    // ---- SAGPool score (fused two score convs) ----
    k_gemv<<<NODES / 8, 256, 0, stream>>>(H2, sWsum, normOut, uvec);
    k_score<<<NODES / 256, 256, 0, stream>>>(uvec, rowStart, degIn, csrSrc, normIn, sb1, sb2, score);

    // ---- per-graph top-K ----
    k_topk<<<BGR, NPG, 0, stream>>>(score, topIdx);

    // ---- gated gather + readout ----
    k_pool<<<(BGR * KKEEP) / 8, 256, 0, stream>>>(H2, score, topIdx, pooled);
    k_readout<<<BGR, 512, 0, stream>>>(pooled, gout);

    (void)in_sizes; (void)n_in; (void)out_size; (void)ws_size;
}
